// Attention_15315853377767
// MI455X (gfx1250) — matrix operand order repaired and checked by reading, not compiled, not run on hardware
//
#include <hip/hip_runtime.h>
#include <hip/hip_bf16.h>
#include <cmath>

// ---------------------------------------------------------------------------
// Types
// ---------------------------------------------------------------------------
typedef _Float16 half4 __attribute__((ext_vector_type(4)));
typedef _Float16 half8 __attribute__((ext_vector_type(8)));
typedef _Float16 v16h  __attribute__((ext_vector_type(16)));
typedef float    v8f   __attribute__((ext_vector_type(8)));

#define B_  2
#define S_  2048
#define D_  2048
#define H_  16
#define HD_ 128

// ---------------------------------------------------------------------------
// WMMA wrapper: D = A(16x32 f16) * B(32x16 f16) + C(16x16 f32)
// ---------------------------------------------------------------------------
__device__ __forceinline__ v8f wmma_f16(v16h a, v16h b, v8f c) {
    return __builtin_amdgcn_wmma_f32_16x16x32_f16(
        /*neg_a=*/false, a, /*neg_b=*/false, b,
        /*c_mod=*/(short)0, c, /*reuse_a=*/false, /*reuse_b=*/false);
}

// ---------------------------------------------------------------------------
// Fragment loaders. Memory is "rows x contiguous-K" (row stride in halves).
// A 16x32 f16: lane = 16*khi + m; element e -> K = (e&8)*2 + (e&7) + 8*khi
// B 32x16 f16: lane = 16*khi + n; element e -> K = (e&8)*2 + (e&7) + 8*khi
// ---------------------------------------------------------------------------
__device__ __forceinline__ v16h load_frag_A(const _Float16* base, int row0,
                                            int k0, int strideHalves) {
    const int lane = threadIdx.x & 31;
    const int row  = row0 + (lane & 15);
    const int khi  = lane >> 4;
    const _Float16* p = base + (size_t)row * strideHalves + k0 + khi * 8;
    half8 lo = *(const half8*)(p);
    half8 hi = *(const half8*)(p + 16);
    return __builtin_shufflevector(lo, hi, 0,1,2,3,4,5,6,7,8,9,10,11,12,13,14,15);
}

__device__ __forceinline__ v16h load_frag_B(const _Float16* base, int row0,
                                            int k0, int strideHalves) {
    const int lane = threadIdx.x & 31;
    const int row  = row0 + (lane & 15);
    const int khi  = lane >> 4;
    const _Float16* p = base + (size_t)row * strideHalves + k0 + khi * 8;
    half8 lo = *(const half8*)(p);
    half8 hi = *(const half8*)(p + 16);
    return __builtin_shufflevector(lo, hi, 0,1,2,3,4,5,6,7,8,9,10,11,12,13,14,15);
}

// ---------------------------------------------------------------------------
// fp32 -> f16 conversion (vectorized x4)
// ---------------------------------------------------------------------------
__global__ void cvt_f32_f16_kernel(const float* __restrict__ in,
                                   _Float16* __restrict__ out, int n4) {
    int i = blockIdx.x * blockDim.x + threadIdx.x;
    const int stride = gridDim.x * blockDim.x;
    for (; i < n4; i += stride) {
        float4 v = ((const float4*)in)[i];
        half4 h;
        h[0] = (_Float16)v.x; h[1] = (_Float16)v.y;
        h[2] = (_Float16)v.z; h[3] = (_Float16)v.w;
        ((half4*)out)[i] = h;
    }
}

// ---------------------------------------------------------------------------
// NT GEMM: C[M,N] = A[M,K] * W[N,K]^T  (f16 in, f32 accumulate)
// Block tile 256x128, BK=32, double-buffered LDS (60 KB), one barrier/step.
// 8 waves as 4(M) x 2(N); each wave computes a 64x64 tile via 4x4 WMMAs:
// 16 WMMAs per 16 ds_load_b128 (1.0 matrix-op per LDS load).
// ---------------------------------------------------------------------------
template <bool OUT_F16>
__global__ __launch_bounds__(256)
void gemm_nt_kernel(const _Float16* __restrict__ A,
                    const _Float16* __restrict__ W,
                    void* __restrict__ Cout,
                    int M, int N, int K) {
    constexpr int LDT = 40;   // padded LDS row stride (halves): 80B = 20 banks
    constexpr int BM  = 256;
    constexpr int BN  = 128;
    __shared__ alignas(16) _Float16 As[2][BM * LDT];  // 40960 B
    __shared__ alignas(16) _Float16 Bs[2][BN * LDT];  // 20480 B

    const int tid = threadIdx.x;
    const int m0  = blockIdx.y * BM;
    const int n0  = blockIdx.x * BN;
    const int w   = tid >> 5;
    const int wm  = w >> 1;       // 0..3 -> rows wm*64
    const int wn  = w & 1;        // 0..1 -> cols wn*64
    const int lane    = tid & 31;
    const int halfSel = lane >> 4;
    const int nIn     = lane & 15;

    const v8f vzero = {};
    v8f acc[4][4];
#pragma unroll
    for (int i = 0; i < 4; ++i)
#pragma unroll
        for (int j = 0; j < 4; ++j) acc[i][j] = vzero;

    // staging: A: one thread per row (64B = 4x b128); B: thread pair per row
    const _Float16* gA = A + (size_t)(m0 + tid) * K;
    const int sbr = tid >> 1;    // 0..127
    const int sbp = tid & 1;     // 0..1
    const _Float16* gB = W + (size_t)(n0 + sbr) * K + sbp * 16;
    const int aoff = tid * LDT;
    const int boff = sbr * LDT + sbp * 16;

    // preload k-slice 0 into buffer 0
#pragma unroll
    for (int q4 = 0; q4 < 4; ++q4)
        *(half8*)&As[0][aoff + q4 * 8] = *(const half8*)(gA + q4 * 8);
    *(half8*)&Bs[0][boff]     = *(const half8*)(gB);
    *(half8*)&Bs[0][boff + 8] = *(const half8*)(gB + 8);
    __syncthreads();

    int buf = 0;
    for (int k0 = 0; k0 < K; k0 += 32, buf ^= 1) {
        // ---- global prefetch of next slice into registers ----
        half8 pa[4] = {};
        half8 pb0 = {}, pb1 = {};
        const bool hasNext = (k0 + 32) < K;
        if (hasNext) {
#pragma unroll
            for (int q4 = 0; q4 < 4; ++q4)
                pa[q4] = *(const half8*)(gA + k0 + 32 + q4 * 8);
            pb0 = *(const half8*)(gB + k0 + 32);
            pb1 = *(const half8*)(gB + k0 + 40);
        }

        // ---- all fragments of this slice up front (16x ds_load_b128) ----
        v16h a0 = load_frag_A(As[buf], wm * 64 +  0, 0, LDT);
        v16h a1 = load_frag_A(As[buf], wm * 64 + 16, 0, LDT);
        v16h a2 = load_frag_A(As[buf], wm * 64 + 32, 0, LDT);
        v16h a3 = load_frag_A(As[buf], wm * 64 + 48, 0, LDT);
        v16h b0 = load_frag_B(Bs[buf], wn * 64 +  0, 0, LDT);
        v16h b1 = load_frag_B(Bs[buf], wn * 64 + 16, 0, LDT);
        v16h b2 = load_frag_B(Bs[buf], wn * 64 + 32, 0, LDT);
        v16h b3 = load_frag_B(Bs[buf], wn * 64 + 48, 0, LDT);

        // ---- 16 WMMAs, independent accumulators ----
        acc[0][0] = wmma_f16(a0, b0, acc[0][0]);
        acc[1][0] = wmma_f16(a1, b0, acc[1][0]);
        acc[2][0] = wmma_f16(a2, b0, acc[2][0]);
        acc[3][0] = wmma_f16(a3, b0, acc[3][0]);
        acc[0][1] = wmma_f16(a0, b1, acc[0][1]);
        acc[1][1] = wmma_f16(a1, b1, acc[1][1]);
        acc[2][1] = wmma_f16(a2, b1, acc[2][1]);
        acc[3][1] = wmma_f16(a3, b1, acc[3][1]);
        acc[0][2] = wmma_f16(a0, b2, acc[0][2]);
        acc[1][2] = wmma_f16(a1, b2, acc[1][2]);
        acc[2][2] = wmma_f16(a2, b2, acc[2][2]);
        acc[3][2] = wmma_f16(a3, b2, acc[3][2]);
        acc[0][3] = wmma_f16(a0, b3, acc[0][3]);
        acc[1][3] = wmma_f16(a1, b3, acc[1][3]);
        acc[2][3] = wmma_f16(a2, b3, acc[2][3]);
        acc[3][3] = wmma_f16(a3, b3, acc[3][3]);

        // ---- commit prefetch to the other buffer ----
        if (hasNext) {
            const int nb = buf ^ 1;
#pragma unroll
            for (int q4 = 0; q4 < 4; ++q4)
                *(half8*)&As[nb][aoff + q4 * 8] = pa[q4];
            *(half8*)&Bs[nb][boff]     = pb0;
            *(half8*)&Bs[nb][boff + 8] = pb1;
        }
        __syncthreads();
    }

    // Epilogue. C layout: lane%16 = N, lane/16 adds 8 to M, vgpr r = M%8.
#pragma unroll
    for (int i = 0; i < 4; ++i)
#pragma unroll
        for (int j = 0; j < 4; ++j)
#pragma unroll
            for (int rr = 0; rr < 8; ++rr) {
                const int row = m0 + wm * 64 + i * 16 + rr + halfSel * 8;
                const int col = n0 + wn * 64 + j * 16 + nIn;
                const float v = acc[i][j][rr];
                if (OUT_F16)
                    ((_Float16*)Cout)[(size_t)row * N + col] = (_Float16)v;
                else
                    ((float*)Cout)[(size_t)row * N + col] = v;
            }
}

// ---------------------------------------------------------------------------
// Flash attention: per block = (b, h, 128 q rows). 8 waves, 16 q rows each.
// KV tiles of 64; online softmax; next tile prefetched into registers so the
// global loads overlap with the 32 WMMAs + softmax of the current tile.
// ---------------------------------------------------------------------------
__global__ __launch_bounds__(256)
void flash_kernel(const _Float16* __restrict__ Q,
                  const _Float16* __restrict__ Km,
                  const _Float16* __restrict__ V,
                  _Float16* __restrict__ O) {
    constexpr int LKT = 136;  // K-tile stride: rows = kv, 128 hd + pad
    constexpr int LVT = 72;   // Vt stride: rows = hd, 64 kv + pad
    constexpr int LPT = 72;   // P stride: rows = q, 64 kv + pad
    __shared__ alignas(16) _Float16 Ks[64 * LKT];
    __shared__ alignas(16) _Float16 Vt[128 * LVT];
    __shared__ alignas(16) _Float16 Ps[8 * 16 * LPT];

    const int tid = threadIdx.x;
    const int qb  = blockIdx.x;
    const int h   = blockIdx.y;
    const int b   = blockIdx.z;
    const int w   = tid >> 5;
    const int lane    = tid & 31;
    const int halfSel = lane >> 4;
    const int nIn     = lane & 15;

    const size_t rowBase = (size_t)b * S_;
    const int hOff = h * HD_;
    const int qr0  = qb * 128 + w * 16;

    // Q fragments (A-style), loaded once from global
    v16h qa[4];
    {
        const _Float16* qp =
            Q + (rowBase + qr0 + nIn) * D_ + hOff + halfSel * 8;
#pragma unroll
        for (int ks = 0; ks < 4; ++ks) {
            half8 lo = *(const half8*)(qp + ks * 32);
            half8 hi = *(const half8*)(qp + ks * 32 + 16);
            qa[ks] = __builtin_shufflevector(lo, hi,
                0,1,2,3,4,5,6,7,8,9,10,11,12,13,14,15);
        }
    }

    const v8f vzero = {};
    float mrow[8], lrow[8];
    v8f o[8];
#pragma unroll
    for (int rr = 0; rr < 8; ++rr) { mrow[rr] = -3.0e38f; lrow[rr] = 0.f; }
#pragma unroll
    for (int t = 0; t < 8; ++t) o[t] = vzero;

    // staging indices: 256 threads -> 64 rows x 4 parts of 32 halves
    const int sr = tid >> 2;     // 0..63 (kv row in tile)
    const int sp = tid & 3;      // 0..3  (32-half column chunk)
    const _Float16* gK = Km + rowBase * D_ + hOff;
    const _Float16* gV = V  + rowBase * D_ + hOff;

    // register-staged tile (software pipeline across the barrier)
    half8 rk[4], rv[4];
    {
        const _Float16* srcK = gK + (size_t)sr * D_ + sp * 32;
        const _Float16* srcV = gV + (size_t)sr * D_ + sp * 32;
#pragma unroll
        for (int q4 = 0; q4 < 4; ++q4) {
            rk[q4] = *(const half8*)(srcK + q4 * 8);
            rv[q4] = *(const half8*)(srcV + q4 * 8);
        }
    }

    for (int kv0 = 0; kv0 < S_; kv0 += 64) {
        __syncthreads();   // previous tile's LDS reads complete
        // commit staged registers to LDS
#pragma unroll
        for (int q4 = 0; q4 < 4; ++q4)
            *(half8*)&Ks[sr * LKT + sp * 32 + q4 * 8] = rk[q4];
#pragma unroll
        for (int q4 = 0; q4 < 4; ++q4)
#pragma unroll
            for (int e = 0; e < 8; ++e)
                Vt[(sp * 32 + q4 * 8 + e) * LVT + sr] = rv[q4][e];
        __syncthreads();

        // prefetch next tile into registers (overlaps with compute below)
        if (kv0 + 64 < S_) {
            const _Float16* srcK = gK + (size_t)(kv0 + 64 + sr) * D_ + sp * 32;
            const _Float16* srcV = gV + (size_t)(kv0 + 64 + sr) * D_ + sp * 32;
#pragma unroll
            for (int q4 = 0; q4 < 4; ++q4) {
                rk[q4] = *(const half8*)(srcK + q4 * 8);
                rv[q4] = *(const half8*)(srcV + q4 * 8);
            }
        }

        // S = Q * K^T : 16 q rows x 64 kv cols per wave
        v8f sacc[4];
#pragma unroll
        for (int j = 0; j < 4; ++j) sacc[j] = vzero;
#pragma unroll
        for (int ks = 0; ks < 4; ++ks) {
            v16h bk0 = load_frag_B(Ks,  0, ks * 32, LKT);
            v16h bk1 = load_frag_B(Ks, 16, ks * 32, LKT);
            v16h bk2 = load_frag_B(Ks, 32, ks * 32, LKT);
            v16h bk3 = load_frag_B(Ks, 48, ks * 32, LKT);
            sacc[0] = wmma_f16(qa[ks], bk0, sacc[0]);
            sacc[1] = wmma_f16(qa[ks], bk1, sacc[1]);
            sacc[2] = wmma_f16(qa[ks], bk2, sacc[2]);
            sacc[3] = wmma_f16(qa[ks], bk3, sacc[3]);
        }

        // online softmax (row = rr + 8*halfSel, spread over 16 lanes)
        _Float16* Pw = &Ps[w * 16 * LPT];
#pragma unroll
        for (int rr = 0; rr < 8; ++rr) {
            float mx = fmaxf(fmaxf(sacc[0][rr], sacc[1][rr]),
                             fmaxf(sacc[2][rr], sacc[3][rr]));
            mx = fmaxf(mx, __shfl_xor(mx, 1, 32));
            mx = fmaxf(mx, __shfl_xor(mx, 2, 32));
            mx = fmaxf(mx, __shfl_xor(mx, 4, 32));
            mx = fmaxf(mx, __shfl_xor(mx, 8, 32));
            const float mnew = fmaxf(mrow[rr], mx);
            const float sc = __expf(mrow[rr] - mnew);
            mrow[rr] = mnew;
            lrow[rr] *= sc;
            float rs = 0.f;
#pragma unroll
            for (int j = 0; j < 4; ++j) {
                const float p = __expf(sacc[j][rr] - mnew);
                rs += p;
                Pw[(rr + halfSel * 8) * LPT + j * 16 + nIn] = (_Float16)p;
            }
            rs += __shfl_xor(rs, 1, 32);
            rs += __shfl_xor(rs, 2, 32);
            rs += __shfl_xor(rs, 4, 32);
            rs += __shfl_xor(rs, 8, 32);
            lrow[rr] += rs;
#pragma unroll
            for (int t = 0; t < 8; ++t) o[t][rr] = o[t][rr] * sc;
        }

        // O += P * V  (P read back as A fragments; per-wave LDS, DS in-order)
#pragma unroll
        for (int ks2 = 0; ks2 < 2; ++ks2) {
            v16h pa = load_frag_A(Pw, 0, ks2 * 32, LPT);
#pragma unroll
            for (int tg = 0; tg < 2; ++tg) {
                v16h v0 = load_frag_B(Vt, tg * 64 +  0, ks2 * 32, LVT);
                v16h v1 = load_frag_B(Vt, tg * 64 + 16, ks2 * 32, LVT);
                v16h v2 = load_frag_B(Vt, tg * 64 + 32, ks2 * 32, LVT);
                v16h v3 = load_frag_B(Vt, tg * 64 + 48, ks2 * 32, LVT);
                o[tg * 4 + 0] = wmma_f16(pa, v0, o[tg * 4 + 0]);
                o[tg * 4 + 1] = wmma_f16(pa, v1, o[tg * 4 + 1]);
                o[tg * 4 + 2] = wmma_f16(pa, v2, o[tg * 4 + 2]);
                o[tg * 4 + 3] = wmma_f16(pa, v3, o[tg * 4 + 3]);
            }
        }
    }

    // finalize and store f16 attention output
#pragma unroll
    for (int rr = 0; rr < 8; ++rr) {
        const float inv = 1.0f / lrow[rr];
        const size_t row = rowBase + qr0 + rr + halfSel * 8;
#pragma unroll
        for (int t = 0; t < 8; ++t)
            O[row * D_ + hOff + t * 16 + nIn] = (_Float16)(o[t][rr] * inv);
    }
}

// ---------------------------------------------------------------------------
// Host side
// ---------------------------------------------------------------------------
extern "C" void kernel_launch(void* const* d_in, const int* in_sizes, int n_in,
                              void* d_out, int out_size, void* d_ws, size_t ws_size,
                              hipStream_t stream) {
    (void)in_sizes; (void)n_in; (void)out_size; (void)ws_size;
    const float* x  = (const float*)d_in[0];
    const float* wq = (const float*)d_in[1];
    const float* wk = (const float*)d_in[2];
    const float* wv = (const float*)d_in[3];
    const float* wo = (const float*)d_in[4];

    const size_t NX = (size_t)B_ * S_ * D_;  // 8,388,608
    const size_t NW = (size_t)D_ * D_;       // 4,194,304

    _Float16* wsh  = (_Float16*)d_ws;
    _Float16* x16  = wsh;
    _Float16* wq16 = x16  + NX;
    _Float16* wk16 = wq16 + NW;
    _Float16* wv16 = wk16 + NW;
    _Float16* wo16 = wv16 + NW;
    _Float16* q16  = wo16 + NW;
    _Float16* k16  = q16  + NX;
    _Float16* v16  = k16  + NX;
    _Float16* a16  = v16  + NX;   // total ~117.5 MB of workspace

    cvt_f32_f16_kernel<<<1024, 256, 0, stream>>>(x,  x16,  (int)(NX / 4));
    cvt_f32_f16_kernel<<<512,  256, 0, stream>>>(wq, wq16, (int)(NW / 4));
    cvt_f32_f16_kernel<<<512,  256, 0, stream>>>(wk, wk16, (int)(NW / 4));
    cvt_f32_f16_kernel<<<512,  256, 0, stream>>>(wv, wv16, (int)(NW / 4));
    cvt_f32_f16_kernel<<<512,  256, 0, stream>>>(wo, wo16, (int)(NW / 4));

    dim3 gg(D_ / 128, (B_ * S_) / 256);  // (16, 16)
    gemm_nt_kernel<true><<<gg, 256, 0, stream>>>(x16, wq16, q16, B_ * S_, D_, D_);
    gemm_nt_kernel<true><<<gg, 256, 0, stream>>>(x16, wk16, k16, B_ * S_, D_, D_);
    gemm_nt_kernel<true><<<gg, 256, 0, stream>>>(x16, wv16, v16, B_ * S_, D_, D_);

    dim3 gf(S_ / 128, H_, B_);           // (16, 16, 2)
    flash_kernel<<<gf, 256, 0, stream>>>(q16, k16, v16, a16);

    gemm_nt_kernel<false><<<gg, 256, 0, stream>>>(a16, wo16, d_out, B_ * S_, D_, D_);
}
